// MultiAgentGRULoop_52355651338937
// MI455X (gfx1250) — matrix operand order repaired and checked by reading, not compiled, not run on hardware
//
#include <hip/hip_runtime.h>

typedef __attribute__((ext_vector_type(16))) _Float16 v16h;
typedef __attribute__((ext_vector_type(8)))  float    v8f;

constexpr int T = 64;        // N_STEP
constexpr int H = 128;       // HID
constexpr int B = 32 * 128;  // N_SC * N_AG = 4096

__device__ __forceinline__ float fast_sigmoid(float x) {
    // v_exp_f32 + v_rcp_f32 (avoid the precise-division expansion)
    return __builtin_amdgcn_rcpf(1.0f + __expf(-x));
}
__device__ __forceinline__ float fast_tanh(float x) {
    // tanh(x) = (1 - e^-2x) / (1 + e^-2x)
    const float e = __expf(-2.0f * x);
    return (1.0f - e) * __builtin_amdgcn_rcpf(1.0f + e);
}

// Swizzle map: hidden index k (0..127) -> (chunk c, lane-half bit, i within 16-half frag)
//   c = k >> 5 ; kk = k & 31
//   halfs 0..7  of a lane hold K = c*32 +      hi*8 + j
//   halfs 8..15 of a lane hold K = c*32 + 16 + hi*8 + j
// so: hi = (kk>>3)&1 ; i = (kk&7) + 8*(kk>>4) ; lane = hi*16 + row

__global__ __launch_bounds__(256)
void gru_layer_wmma(const float* in, float* out,
                    const float* __restrict__ w_i, const float* __restrict__ w_h,
                    const float* __restrict__ b_i, const float* __restrict__ b_h,
                    const unsigned char* __restrict__ inval,
                    int last_layer) {
    // A-operand tiles stored pre-swizzled: [chunk][lane][16 halfs] (32B per lane, contiguous)
    __shared__ __align__(32) _Float16 a_x[4][32][16];
    __shared__ __align__(32) _Float16 a_h[4][32][16];
    __shared__ unsigned char inv_lds[16];

    const int tid  = threadIdx.x;
    const int lane = tid & 31;
    const int wave = tid >> 5;      // 0..7 -> 16-column group in every gate
    const int hi   = lane >> 4;
    const int col  = lane & 15;
    const int b0   = blockIdx.x * 16;

    // ---- Load weights once into VGPR fragments (B-operand, 32x16 f16). ----
    // lane holds column n = g*128 + 16*wave + col; halfs i<8 are K = c*32 + hi*8 + i, i>=8 are K = c*32 + 16 + hi*8 + (i-8).
    v16h wif[3][4], whf[3][4];
    float bsum[2], bin, bhn;
#pragma unroll
    for (int g = 0; g < 3; ++g) {
        const int n = g * 128 + wave * 16 + col;
#pragma unroll
        for (int c = 0; c < 4; ++c) {
            const int kb = c * 32 + hi * 8;
            const float4* pi = (const float4*)&w_i[n * H + kb];
            const float4* ph = (const float4*)&w_h[n * H + kb];
            v16h fi, fh;
#pragma unroll
            for (int q = 0; q < 4; ++q) {
                float4 vi = pi[q + (q & 2)], vh = ph[q + (q & 2)];
                fi[q * 4 + 0] = (_Float16)vi.x; fi[q * 4 + 1] = (_Float16)vi.y;
                fi[q * 4 + 2] = (_Float16)vi.z; fi[q * 4 + 3] = (_Float16)vi.w;
                fh[q * 4 + 0] = (_Float16)vh.x; fh[q * 4 + 1] = (_Float16)vh.y;
                fh[q * 4 + 2] = (_Float16)vh.z; fh[q * 4 + 3] = (_Float16)vh.w;
            }
            wif[g][c] = fi;
            whf[g][c] = fh;
        }
    }
    {
        const int nc = wave * 16 + col;
        bsum[0] = b_i[0 * 128 + nc] + b_h[0 * 128 + nc];   // r
        bsum[1] = b_i[1 * 128 + nc] + b_h[1 * 128 + nc];   // z
        bin     = b_i[2 * 128 + nc];                        // n (input side)
        bhn     = b_h[2 * 128 + nc];                        // n (hidden side)
    }

    // ---- h0 = 0 (LDS swizzled copy + register copy) ----
    {
        uint4 z{0, 0, 0, 0};
        uint4* p = (uint4*)a_h;              // 4*32*16 halfs = 4096 B = 256 uint4
        p[tid] = z;
    }
    float hreg[8];
#pragma unroll
    for (int e = 0; e < 8; ++e) hreg[e] = 0.0f;

    // h-write swizzle constants (k = wave*16 + col is fixed per lane)
    const int kg   = wave * 16 + col;
    const int c2   = kg >> 5;
    const int kk2  = kg & 31;
    const int hwl  = ((kk2 >> 3) & 1) << 4;            // lane-half bit of dest lane
    const int i2   = (kk2 & 7) + 8 * (kk2 >> 4);       // frag element index

    // x-staging constants: thread owns row r, 8 consecutive k at k0
    const int sr = tid >> 4;
    const int k0 = (tid & 15) * 8;
    const int sc = k0 >> 5, skk = k0 & 31;
    const int sl = (((skk >> 3) & 1) << 4) | sr;
    const int si = (skk >> 4) * 8;
    uint4* xdst = (uint4*)&a_x[sc][sl][si];            // 16B-aligned (si in {0,8})

    for (int t = 0; t < T; ++t) {
        // ---- stage input tile (f32 global -> f16 swizzled LDS) ----
        {
            const float4* src = (const float4*)&in[((size_t)(b0 + sr) * T + t) * H + k0];
            float4 f0 = src[0], f1 = src[1];
            union { _Float16 h[8]; uint4 u; } pk;
            pk.h[0] = (_Float16)f0.x; pk.h[1] = (_Float16)f0.y;
            pk.h[2] = (_Float16)f0.z; pk.h[3] = (_Float16)f0.w;
            pk.h[4] = (_Float16)f1.x; pk.h[5] = (_Float16)f1.y;
            pk.h[6] = (_Float16)f1.z; pk.h[7] = (_Float16)f1.w;
            *xdst = pk.u;
        }
        if (tid < 16)
            inv_lds[tid] = inval[(size_t)(b0 + tid) * T + t];
        if (t + 1 < T)
            __builtin_prefetch(&in[((size_t)(b0 + sr) * T + (t + 1)) * H + k0], 0, 1);
        __syncthreads();   // staging + previous-step h writes visible

        // ---- gi = x@Wi^T ; gh = h@Wh^T  (inline-0 C, biases folded into gate math) ----
        v8f acc_i[3] = {v8f{0}, v8f{0}, v8f{0}};
        v8f acc_h[3] = {v8f{0}, v8f{0}, v8f{0}};
#pragma unroll
        for (int c = 0; c < 4; ++c) {
            v16h ax = *(const v16h*)&a_x[c][lane][0];
            v16h ah = *(const v16h*)&a_h[c][lane][0];
#pragma unroll
            for (int g = 0; g < 3; ++g) {
                acc_i[g] = __builtin_amdgcn_wmma_f32_16x16x32_f16(
                    false, ax, false, wif[g][c], (short)0, acc_i[g], false, false);
                acc_h[g] = __builtin_amdgcn_wmma_f32_16x16x32_f16(
                    false, ah, false, whf[g][c], (short)0, acc_h[g], false, false);
            }
        }

        // ---- gates + state update; C/D tile rows e+8*hi are exactly this lane's h rows ----
        float hnew[8], outv[8];
#pragma unroll
        for (int e = 0; e < 8; ++e) {
            const int rr = e + 8 * hi;
            const float rg = fast_sigmoid(acc_i[0][e] + acc_h[0][e] + bsum[0]);
            const float zg = fast_sigmoid(acc_i[1][e] + acc_h[1][e] + bsum[1]);
            const float ng = fast_tanh(acc_i[2][e] + bin + rg * (acc_h[2][e] + bhn));
            const float hv = (1.0f - zg) * ng + zg * hreg[e];
            const bool  m  = inv_lds[rr] != 0;
            hnew[e] = m ? 0.0f : hv;                  // carried h is masked
            outv[e] = (last_layer && m) ? 0.0f : hv;  // inter-layer out unmasked
            hreg[e] = hnew[e];
        }
        __syncthreads();   // all A-fragment reads of this step done

        // ---- write h back to swizzled LDS + output to global ----
#pragma unroll
        for (int e = 0; e < 8; ++e) {
            const int rr = e + 8 * hi;
            a_h[c2][hwl | rr][i2] = (_Float16)hnew[e];
            out[((size_t)(b0 + rr) * T + t) * H + kg] = outv[e];
        }
        // next iteration's staging writes a_x only; a_h reads are fenced by the top barrier
    }
}

extern "C" void kernel_launch(void* const* d_in, const int* in_sizes, int n_in,
                              void* d_out, int out_size, void* d_ws, size_t ws_size,
                              hipStream_t stream) {
    const float*         x     = (const float*)d_in[0];
    const unsigned char* inval = (const unsigned char*)d_in[1]; // bool [B,T]
    const float*         w_ih  = (const float*)d_in[2];         // [3,384,128]
    const float*         w_hh  = (const float*)d_in[3];
    const float*         b_ih  = (const float*)d_in[4];         // [3,384]
    const float*         b_hh  = (const float*)d_in[5];
    float*               out   = (float*)d_out;                 // [B,T,H] == [sc,ag,t,h]

    dim3 grid(B / 16), block(256);
    for (int l = 0; l < 3; ++l) {
        const float* in = (l == 0) ? x : out;   // layers 1,2 run in place on d_out
        gru_layer_wmma<<<grid, block, 0, stream>>>(
            in, out,
            w_ih + (size_t)l * 384 * 128, w_hh + (size_t)l * 384 * 128,
            b_ih + (size_t)l * 384,       b_hh + (size_t)l * 384,
            inval, (l == 2) ? 1 : 0);
    }
}
